// GNN_26053271618008
// MI455X (gfx1250) — matrix operand order repaired and checked by reading, not compiled, not run on hardware
//
#include <hip/hip_runtime.h>
#include <math.h>

#define NN 4096
#define NE 8192
#define PI_F  3.14159265358979f
#define TPI_F 6.28318530717959f

typedef _Float16 v16h __attribute__((ext_vector_type(16)));
typedef float    v8f  __attribute__((ext_vector_type(8)));

struct Rho { float a, b, c; };   // symmetric 2x2 density matrix [[a,b],[b,c]]

__device__ __forceinline__ Rho rho_init(float ang) {
  float s, c;
  __sincosf(0.5f * ang, &s, &c);
  Rho r; r.a = c * c; r.b = c * s; r.c = s * s;
  return r;
}

// rho[t] = Tr_ctrl( CNOT (RY(thc) rho_c RY^T  kron  RY(tht) rho_t RY^T) CNOT )
//        = p0 * rt + p1 * (X rt X)   (exact; all matrices real symmetric)
__device__ __forceinline__ void tstep(Rho* rho, int ci, int ti, float thc, float tht) {
  float sc, cc; __sincosf(0.5f * thc, &sc, &cc);
  Rho rc = rho[ci];
  float p0 = cc*cc*rc.a - 2.f*cc*sc*rc.b + sc*sc*rc.c;
  float p1 = sc*sc*rc.a + 2.f*cc*sc*rc.b + cc*cc*rc.c;
  float st, ct; __sincosf(0.5f * tht, &st, &ct);
  Rho rt = rho[ti];
  float t00 = ct*ct*rt.a - 2.f*ct*st*rt.b + st*st*rt.c;
  float t01 = ct*st*(rt.a - rt.c) + (ct*ct - st*st)*rt.b;
  float t11 = st*st*rt.a + 2.f*ct*st*rt.b + ct*ct*rt.c;
  Rho o;
  o.a = p0*t00 + p1*t11;
  o.b = (p0 + p1)*t01;
  o.c = p0*t11 + p1*t00;
  rho[ti] = o;
}

__device__ __forceinline__ void conj2(float th, Rho r, float M[2][2]) {
  float s, c; __sincosf(0.5f * th, &s, &c);
  M[0][0] = c*c*r.a - 2.f*c*s*r.b + s*s*r.c;
  M[0][1] = c*s*(r.a - r.c) + (c*c - s*s)*r.b;
  M[1][0] = M[0][1];
  M[1][1] = s*s*r.a + 2.f*c*s*r.b + c*c*r.c;
}

// ---- recover send[]/recv[] from the dense one-hot incidence matrices (one coalesced pass)
__global__ void k_extract(const float* __restrict__ Ri, const float* __restrict__ Ro,
                          int* __restrict__ send, int* __restrict__ recv) {
  int e = blockIdx.x * blockDim.x + threadIdx.x;
  if (e >= NE) return;
  int s = 0, r = 0;
  for (int n = 0; n < NN; ++n) {
    if (Ro[(size_t)n * NE + e] > 0.5f) s = n;
    if (Ri[(size_t)n * NE + e] > 0.5f) r = n;
  }
  send[e] = s; recv[e] = r;
}

// ---- InputNet: H = [2pi*sigmoid(X W1 + b1), X]
__global__ void k_input(const float* __restrict__ X, const float* __restrict__ W1,
                        const float* __restrict__ b1, float* __restrict__ H) {
  int n = blockIdx.x * blockDim.x + threadIdx.x;
  if (n >= NN) return;
  float x0 = X[n*3+0], x1 = X[n*3+1], x2 = X[n*3+2];
  #pragma unroll
  for (int j = 0; j < 5; ++j) {
    float t = x0*W1[0*5+j] + x1*W1[1*5+j] + x2*W1[2*5+j] + b1[j];
    H[n*8+j] = TPI_F / (1.f + __expf(-t));
  }
  H[n*8+5] = x0; H[n*8+6] = x1; H[n*8+7] = x2;
}

// ---- edge QNN (15 tree steps on 16 wires), gathers H rows via indices
__global__ void k_edge(const float* __restrict__ H, const int* __restrict__ send,
                       const int* __restrict__ recv, const float* __restrict__ th,
                       float* __restrict__ out) {
  int e = blockIdx.x * blockDim.x + threadIdx.x;
  if (e >= NE) return;
  int s = send[e], r = recv[e];
  Rho rho[16];
  #pragma unroll
  for (int j = 0; j < 8; ++j) rho[j]     = rho_init(H[s*8+j]);
  #pragma unroll
  for (int j = 0; j < 8; ++j) rho[8 + j] = rho_init(H[r*8+j]);
  const int ES[15][4] = {
    {0,1,0,1},{3,2,3,2},{4,5,4,5},{7,6,7,6},{8,9,8,9},{11,10,11,10},
    {12,13,12,13},{15,14,15,14},{1,2,16,17},{6,5,19,18},{9,10,20,21},
    {14,13,23,22},{2,5,24,25},{13,10,27,26},{5,10,28,29}};
  #pragma unroll
  for (int k = 0; k < 15; ++k)
    tstep(rho, ES[k][0], ES[k][1], th[ES[k][2]], th[ES[k][3]]);
  float M[2][2];
  conj2(th[30], rho[10], M);
  out[e] = 0.5f * (1.f - (M[0][0] - M[1][1]));
}

__global__ void k_zero(float* __restrict__ p, int n) {
  int i = blockIdx.x * blockDim.x + threadIdx.x;
  if (i < n) p[i] = 0.f;
}

// ---- mi[n] = sum_{recv[e]=n} ev[e]*H[send[e]] ; mo[n] = sum_{send[e]=n} ev[e]*H[recv[e]]
__global__ void k_scatter(const float* __restrict__ H, const float* __restrict__ ev,
                          const int* __restrict__ send, const int* __restrict__ recv,
                          float* __restrict__ mi, float* __restrict__ mo) {
  int e = blockIdx.x * blockDim.x + threadIdx.x;
  if (e >= NE) return;
  int s = send[e], r = recv[e];
  float w = ev[e];
  #pragma unroll
  for (int j = 0; j < 8; ++j) {
    atomicAdd(&mi[r*8+j], w * H[s*8+j]);
    atomicAdd(&mo[s*8+j], w * H[r*8+j]);
  }
}

// ---- one-time: build U (32x32) from the 11-gate tail, then G_p = U^T diag(signs_p) U
__device__ __forceinline__ void do_rot(float (*dst)[32], const float (*src)[32],
                                       const float* thn, int t0, int t1, int t2, int t3, int t4,
                                       int i, int j) {
  int tt[5] = {t0, t1, t2, t3, t4};
  float cc[5], ss[5];
  #pragma unroll
  for (int p = 0; p < 5; ++p) {
    if (tt[p] >= 0) __sincosf(0.5f * thn[tt[p]], &ss[p], &cc[p]);
    else { cc[p] = 1.f; ss[p] = 0.f; }
  }
  float acc = 0.f;
  for (int k = 0; k < 32; ++k) {
    float w = 1.f;
    #pragma unroll
    for (int p = 0; p < 5; ++p) {
      int ri = (i >> (4 - p)) & 1, ciq = (k >> (4 - p)) & 1;
      w *= ri ? (ciq ? cc[p] : ss[p]) : (ciq ? -ss[p] : cc[p]);
    }
    acc += w * src[k][j];
  }
  dst[i][j] = acc;
}

__device__ __forceinline__ void do_cnot(float (*dst)[32], const float (*src)[32],
                                        int cw, int tw, int i, int j) {
  int g = ((i >> (4 - cw)) & 1) ? (i ^ (1 << (4 - tw))) : i;
  dst[i][j] = src[g][j];
}

__global__ void k_precompute(const float* __restrict__ thn, _Float16* __restrict__ Gmat) {
  __shared__ float Ua[32][32];
  __shared__ float Ub[32][32];
  int tid = threadIdx.x;
  int i = tid >> 5, j = tid & 31;
  Ua[i][j] = (i == j) ? 1.f : 0.f;
  __syncthreads();
  do_rot(Ub, Ua, thn, -1,42,43,-1,-1, i, j); __syncthreads();
  do_cnot(Ua, Ub, 1, 2, i, j);               __syncthreads();
  do_rot(Ub, Ua, thn, -1,-1,44,45,-1, i, j); __syncthreads();
  do_cnot(Ua, Ub, 2, 3, i, j);               __syncthreads();
  do_rot(Ub, Ua, thn, -1,46,47,-1,-1, i, j); __syncthreads();
  do_cnot(Ua, Ub, 2, 1, i, j);               __syncthreads();
  do_rot(Ub, Ua, thn, -1,-1,-1,48,49, i, j); __syncthreads();
  do_cnot(Ua, Ub, 3, 4, i, j);               __syncthreads();
  do_rot(Ub, Ua, thn, 50,51,-1,-1,-1, i, j); __syncthreads();
  do_cnot(Ua, Ub, 1, 0, i, j);               __syncthreads();
  do_rot(Ub, Ua, thn, 52,53,54,55,56, i, j); __syncthreads();
  // Gmat[p][i*32+j] = sum_k U[k][i]*sgn(p,k)*U[k][j] ; rows 5..15 zero-padded (f16, |G|<=1)
  for (int idx = tid; idx < 16 * 1024; idx += 1024) {
    int p = idx >> 10, q = idx & 1023, ii = q >> 5, jj = q & 31;
    float v = 0.f;
    if (p < 5) {
      for (int k = 0; k < 32; ++k) {
        float sgn = ((k >> (4 - p)) & 1) ? -1.f : 1.f;
        v += sgn * Ub[k][ii] * Ub[k][jj];
      }
    }
    Gmat[idx] = (_Float16)v;
  }
}

// ---- node QNN: one wave = 16 nodes. Scalar 19-step circuit -> R tile (16x1024 f16) in LDS,
//      then Z = R * G^T via 32x v_wmma_f32_16x16x32_f16 ; lanes n=5..7 write the X columns.
__global__ void __launch_bounds__(32) k_node(
    const float* __restrict__ mi, const float* __restrict__ mo,
    const float* __restrict__ X, const float* __restrict__ thn,
    const _Float16* __restrict__ Gmat, float* __restrict__ H) {
  __shared__ _Float16 Rlds[16 * 1024];
  int lane = threadIdx.x;
  int m = lane & 15, hi = lane >> 4;
  int node = blockIdx.x * 16 + m;

  Rho rho[24];
  #pragma unroll
  for (int j = 0; j < 8; ++j) {
    rho[j]      = rho_init(mi[node*8+j]);
    rho[8 + j]  = rho_init(mo[node*8+j]);
    rho[16 + j] = rho_init(H[node*8+j]);
  }
  const int NS[19][4] = {
    {0,1,0,1},{3,2,3,2},{4,5,4,5},{7,6,7,6},{8,9,8,9},{11,10,11,10},
    {12,13,12,13},{15,14,15,14},{16,17,16,17},{19,18,19,18},{20,21,20,21},
    {23,22,23,22},{1,2,24,25},{6,5,27,26},{9,10,28,29},{14,13,31,30},
    {17,18,32,33},{22,21,35,34},{13,10,39,38}};
  #pragma unroll
  for (int k = 0; k < 19; ++k)
    tstep(rho, NS[k][0], NS[k][1], thn[NS[k][2]], thn[NS[k][3]]);

  float A2[2][2], A5[2][2], B18[2][2], B21[2][2];
  conj2(thn[36], rho[2],  A2);
  conj2(thn[37], rho[5],  A5);
  conj2(thn[40], rho[18], B18);
  conj2(thn[41], rho[21], B21);
  float B2[2][2] = {{rho[10].a, rho[10].b}, {rho[10].b, rho[10].c}};

  float A[4][4], C[4][4];
  #pragma unroll
  for (int x = 0; x < 2; ++x)
  #pragma unroll
  for (int a = 0; a < 2; ++a)
  #pragma unroll
  for (int y = 0; y < 2; ++y)
  #pragma unroll
  for (int b = 0; b < 2; ++b) {
    A[x*2+a][y*2+b] = A2[x][y] * A5[a ^ x][b ^ y];    // CNOT(ctrl=msb, targ=lsb) conj
    C[a*2+x][b*2+y] = B18[a ^ x][b ^ y] * B21[x][y];  // CNOT(ctrl=lsb, targ=msb) conj
  }

  // R tile: R[m][i*32+j] = A[ai][aj]*B2[ci][cj]*C[ei][ej], i=(ai<<3)|(ci<<2)|ei
  for (int i = hi * 16; i < hi * 16 + 16; ++i) {
    int ai = i >> 3, ci2 = (i >> 2) & 1, ei = i & 3;
    #pragma unroll
    for (int j = 0; j < 32; ++j) {
      int aj = j >> 3, cj = (j >> 2) & 1, ej = j & 3;
      Rlds[m * 1024 + i * 32 + j] = (_Float16)(A[ai][aj] * B2[ci2][cj] * C[ei][ej]);
    }
  }
  __syncthreads();

  union Frag { unsigned int u[8]; v16h h; };
  v8f acc = {0.f, 0.f, 0.f, 0.f, 0.f, 0.f, 0.f, 0.f};
  for (int kb = 0; kb < 32; ++kb) {
    int kbase = kb * 32;
    Frag fa, fb;
    #pragma unroll
    for (int v = 0; v < 8; ++v) {
      int ka  = kbase + ((v & 3) << 1) + ((v >> 2) << 4) + (hi << 3); // A 16x32 f16 layout
      int kbb = kbase + ((v & 3) << 1) + ((v >> 2) << 4) + (hi << 3); // B 32x16 f16 layout
      fa.u[v] = *(const unsigned int*)&Rlds[m * 1024 + ka];
      fb.u[v] = *(const unsigned int*)(Gmat + m * 1024 + kbb);        // B[k][n=m] = G[n][k]
    }
    acc = __builtin_amdgcn_wmma_f32_16x16x32_f16(false, fa.h, false, fb.h,
                                                 (short)0, acc, false, false);
  }

  #pragma unroll
  for (int g = 0; g < 8; ++g) {
    int nd = blockIdx.x * 16 + g + hi * 8;           // D row -> node
    if (m < 5)      H[nd*8 + m] = PI_F * (1.f - acc[g]);   // 2pi*(1-z)/2
    else if (m < 8) H[nd*8 + m] = X[nd*3 + (m - 5)];       // concat(.., X)
  }
}

extern "C" void kernel_launch(void* const* d_in, const int* in_sizes, int n_in,
                              void* d_out, int out_size, void* d_ws, size_t ws_size,
                              hipStream_t stream) {
  (void)in_sizes; (void)n_in; (void)out_size; (void)ws_size;
  const float* X   = (const float*)d_in[0];
  const float* Ri  = (const float*)d_in[1];
  const float* Ro  = (const float*)d_in[2];
  const float* W1  = (const float*)d_in[3];
  const float* b1  = (const float*)d_in[4];
  const float* thE = (const float*)d_in[5];
  const float* thN = (const float*)d_in[6];
  float* out = (float*)d_out;

  char* ws = (char*)d_ws;
  size_t off = 0;
  auto carve = [&](size_t bytes) -> void* {
    void* p = ws + off;
    off += (bytes + 255) & ~(size_t)255;
    return p;
  };
  int*      send = (int*)     carve(NE * sizeof(int));
  int*      recv = (int*)     carve(NE * sizeof(int));
  float*    H    = (float*)   carve((size_t)NN * 8 * sizeof(float));
  float*    ev   = (float*)   carve(NE * sizeof(float));
  float*    mimo = (float*)   carve((size_t)2 * NN * 8 * sizeof(float));
  float*    mi   = mimo;
  float*    mo   = mimo + (size_t)NN * 8;
  _Float16* Gmat = (_Float16*)carve((size_t)16 * 1024 * sizeof(_Float16));

  k_extract<<<NE/256, 256, 0, stream>>>(Ri, Ro, send, recv);
  k_precompute<<<1, 1024, 0, stream>>>(thN, Gmat);
  k_input<<<NN/256, 256, 0, stream>>>(X, W1, b1, H);
  for (int it = 0; it < 2; ++it) {
    k_edge<<<NE/256, 256, 0, stream>>>(H, send, recv, thE, ev);
    k_zero<<<(2*NN*8)/256, 256, 0, stream>>>(mimo, 2*NN*8);
    k_scatter<<<NE/256, 256, 0, stream>>>(H, ev, send, recv, mi, mo);
    k_node<<<NN/16, 32, 0, stream>>>(mi, mo, X, thN, Gmat, H);
  }
  k_edge<<<NE/256, 256, 0, stream>>>(H, send, recv, thE, out);
}
